// GNN_41300405518878
// MI455X (gfx1250) — compile-verified, matrix operand order checked by reading, not run on hardware
//
#include <hip/hip_runtime.h>
#include <hip/hip_bf16.h>

typedef __attribute__((ext_vector_type(16))) __bf16 v16bf;
typedef __attribute__((ext_vector_type(8)))  float  v8f;

// ---------------------------------------------------------------------------
// Weight repack: W[K,Ncol] f32 row-major -> bf16 fragment-native layout
// Wp[((t*NT + j)*32 + lane)*16 + i]  where for lane L (half=L/16):
//   i in [0,8):  k = t*32 + half*8 + i
//   i in [8,16): k = t*32 + 16 + half*8 + (i-8)
//   n = j*16 + (L&15)
// so a wave's B fragment for (t,j) is 32 lanes x 16 contiguous bf16 (32B/lane).
// ---------------------------------------------------------------------------
__global__ void repack_w_kernel(const float* __restrict__ W, __bf16* __restrict__ Wp,
                                int K, int Ncol, int K32, int ntiles) {
    int idx = blockIdx.x * blockDim.x + threadIdx.x;
    int total = K32 * ntiles * 32 * 16;
    if (idx >= total) return;
    int i    = idx & 15;
    int lane = (idx >> 4) & 31;
    int tj   = idx >> 9;
    int j    = tj % ntiles;
    int t    = tj / ntiles;
    int half = lane >> 4;
    int k = t * 32 + (i < 8 ? half * 8 + i : 16 + half * 8 + (i - 8));
    int n = j * 16 + (lane & 15);
    float v = (k < K && n < Ncol) ? W[(size_t)k * Ncol + n] : 0.0f;
    Wp[idx] = (__bf16)v;
}

// ---------------------------------------------------------------------------
// Degree / norm precompute
// ---------------------------------------------------------------------------
__global__ void init_deg_kernel(float* __restrict__ deg, int n) {
    int i = blockIdx.x * blockDim.x + threadIdx.x;
    if (i < n) deg[i] = 1.0f;   // self-loop contribution
}

__global__ void deg_accum_kernel(const int* __restrict__ dst, float* __restrict__ deg, int E) {
    int e = blockIdx.x * blockDim.x + threadIdx.x;
    if (e < E) atomicAdd(&deg[dst[e]], 1.0f);
}

__global__ void rsqrt_kernel(float* __restrict__ d, int n) {
    int i = blockIdx.x * blockDim.x + threadIdx.x;
    if (i < n) d[i] = __frsqrt_rn(d[i]);   // deg >= 1, safe
}

__global__ void edge_norm_kernel(const int* __restrict__ src, const int* __restrict__ dst,
                                 const float* __restrict__ dis, float* __restrict__ norm, int E) {
    int e = blockIdx.x * blockDim.x + threadIdx.x;
    if (e < E) norm[e] = dis[src[e]] * dis[dst[e]];
}

// ---------------------------------------------------------------------------
// GEMM: C[M,ldc] (f32) = A[M,lda] (f32 -> bf16 in-register) x Wp (bf16 packed)
// 8 waves/block, 1 wave = 16-row strip, NT 16x16 tiles/wave.
// B fragments batched into registers (one load clause, one wait, NT wmma);
// A fragment software-pipelined one K-step ahead.
// ---------------------------------------------------------------------------
template <int NT>
__global__ void __launch_bounds__(256)
gemm_wmma_bf16(const float* __restrict__ A, const __bf16* __restrict__ Wp,
               float* __restrict__ C, int M, int lda, int K, int K32,
               int ldc, int ncolReal) {
    const int lane   = threadIdx.x & 31;
    const int wave   = threadIdx.x >> 5;
    const int mtile  = blockIdx.x * 8 + wave;
    const int mtiles = (M + 15) >> 4;
    if (mtile >= mtiles) return;          // wave-uniform exit: EXEC stays all-1s

    const int half = lane >> 4;
    int row  = mtile * 16 + (lane & 15);
    int rowc = row < M ? row : M - 1;     // clamp (stores are guarded)
    const float* Arow = A + (size_t)rowc * lda;

    auto loadA = [&](int t, v16bf& a) {
        const int kbase = t * 32;
        const int k0 = kbase + half * 8;
        if (kbase + 32 <= K) {
            // interior tile: 4 x 128-bit coalesced loads, pack to bf16
            float4 f0 = *reinterpret_cast<const float4*>(Arow + k0);
            float4 f1 = *reinterpret_cast<const float4*>(Arow + k0 + 4);
            float4 f2 = *reinterpret_cast<const float4*>(Arow + k0 + 16);
            float4 f3 = *reinterpret_cast<const float4*>(Arow + k0 + 20);
            a[0]  = (__bf16)f0.x; a[1]  = (__bf16)f0.y; a[2]  = (__bf16)f0.z; a[3]  = (__bf16)f0.w;
            a[4]  = (__bf16)f1.x; a[5]  = (__bf16)f1.y; a[6]  = (__bf16)f1.z; a[7]  = (__bf16)f1.w;
            a[8]  = (__bf16)f2.x; a[9]  = (__bf16)f2.y; a[10] = (__bf16)f2.z; a[11] = (__bf16)f2.w;
            a[12] = (__bf16)f3.x; a[13] = (__bf16)f3.y; a[14] = (__bf16)f3.z; a[15] = (__bf16)f3.w;
        } else {
            // K boundary tile (K=500): guarded scalar loads, zero-pad
#pragma unroll
            for (int i = 0; i < 16; ++i) {
                int k = k0 + (i < 8 ? i : 8 + i);   // i>=8 -> +16 + (i-8)
                float f = (k < K) ? Arow[k] : 0.0f;
                a[i] = (__bf16)f;
            }
        }
    };

    v8f acc[NT];
    const v8f z8 = {0.f, 0.f, 0.f, 0.f, 0.f, 0.f, 0.f, 0.f};
#pragma unroll
    for (int j = 0; j < NT; ++j) acc[j] = z8;

    v16bf a, an;
    loadA(0, a);
    for (int t = 0; t < K32; ++t) {
        // batch-load all NT B fragments (32B each) -> one clause + one wait
        const __bf16* Wt = Wp + ((size_t)(t * NT) * 32 + lane) * 16;
        v16bf bf[NT];
#pragma unroll
        for (int j = 0; j < NT; ++j)
            bf[j] = *reinterpret_cast<const v16bf*>(Wt + (size_t)j * 32 * 16);
        // prefetch next A fragment under the WMMA block
        if (t + 1 < K32) loadA(t + 1, an);
#pragma unroll
        for (int j = 0; j < NT; ++j)
            acc[j] = __builtin_amdgcn_wmma_f32_16x16x32_bf16(
                false, a, false, bf[j], (short)0, acc[j], false, false);
        a = an;
    }

    // C/D fragment layout: VGPR r -> row (r + 8*half), col = j*16 + (lane&15)
#pragma unroll
    for (int j = 0; j < NT; ++j) {
        int col = j * 16 + (lane & 15);
        if (col >= ncolReal) continue;
#pragma unroll
        for (int r = 0; r < 8; ++r) {
            int rr = mtile * 16 + r + half * 8;
            if (rr < M) C[(size_t)rr * ldc + col] = acc[j][r];
        }
    }
}

// ---------------------------------------------------------------------------
// agg zeroing (b128), edge scatter-add (L2-resident f32 atomics), finalize
// ---------------------------------------------------------------------------
__global__ void zero4_kernel(float4* __restrict__ p, int n4) {
    int i = blockIdx.x * blockDim.x + threadIdx.x;
    if (i < n4) p[i] = make_float4(0.f, 0.f, 0.f, 0.f);
}

__global__ void scatter_edges_kernel(const float* __restrict__ hpre,
                                     const float* __restrict__ norm,
                                     const int* __restrict__ src,
                                     const int* __restrict__ dst,
                                     float* __restrict__ agg, int E) {
    int tid = blockIdx.x * blockDim.x + threadIdx.x;   // E*32 threads, 4 feats each
    int e = tid >> 5;
    if (e >= E) return;
    int c = tid & 31;
    int s = src[e], d = dst[e];
    float nv = norm[e];
    float4 v = *reinterpret_cast<const float4*>(hpre + (size_t)s * 128 + c * 4);
    float* out = agg + (size_t)d * 128 + c * 4;
    atomicAdd(out + 0, nv * v.x);
    atomicAdd(out + 1, nv * v.y);
    atomicAdd(out + 2, nv * v.z);
    atomicAdd(out + 3, nv * v.w);
}

__global__ void finalize_kernel(const float4* __restrict__ agg, const float4* __restrict__ hpre,
                                const float* __restrict__ dis, const float* __restrict__ b,
                                float4* __restrict__ X, int N) {
    int tid = blockIdx.x * blockDim.x + threadIdx.x;   // N*32 threads, 4 feats each
    if (tid >= N * 32) return;
    int i = tid >> 5;
    int c = tid & 31;
    float di = dis[i];
    float d2 = di * di;
    float4 ag = agg[tid];
    float4 hp = hpre[tid];
    float4 bb = *reinterpret_cast<const float4*>(b + c * 4);
    float4 o;
    o.x = fmaxf(ag.x + d2 * hp.x + bb.x, 0.f);
    o.y = fmaxf(ag.y + d2 * hp.y + bb.y, 0.f);
    o.z = fmaxf(ag.z + d2 * hp.z + bb.z, 0.f);
    o.w = fmaxf(ag.w + d2 * hp.w + bb.w, 0.f);
    X[tid] = o;
}

// ---------------------------------------------------------------------------
// Per-node 40-way log_softmax (logits strided by 48 -> 10 x b128 loads, + bias)
// ---------------------------------------------------------------------------
__global__ void logsoftmax_kernel(const float* __restrict__ logits, const float* __restrict__ bfc,
                                  float* __restrict__ out, int N) {
    int i = blockIdx.x * blockDim.x + threadIdx.x;
    if (i >= N) return;
    float v[40];
#pragma unroll
    for (int q = 0; q < 10; ++q) {
        float4 f = *reinterpret_cast<const float4*>(logits + (size_t)i * 48 + q * 4);
        v[q * 4 + 0] = f.x; v[q * 4 + 1] = f.y; v[q * 4 + 2] = f.z; v[q * 4 + 3] = f.w;
    }
    float mx = -3.4e38f;
#pragma unroll
    for (int j = 0; j < 40; ++j) {
        v[j] += bfc[j];
        mx = fmaxf(mx, v[j]);
    }
    float s = 0.0f;
#pragma unroll
    for (int j = 0; j < 40; ++j) s += __expf(v[j] - mx);
    float ls = __logf(s);
#pragma unroll
    for (int j = 0; j < 40; ++j) out[(size_t)i * 40 + j] = v[j] - mx - ls;
}

// ---------------------------------------------------------------------------
// Host launch
// ---------------------------------------------------------------------------
static inline int cdiv(int a, int b) { return (a + b - 1) / b; }

extern "C" void kernel_launch(void* const* d_in, const int* in_sizes, int n_in,
                              void* d_out, int out_size, void* d_ws, size_t ws_size,
                              hipStream_t stream) {
    const int N = in_sizes[0] / 500;
    const int E = in_sizes[1] / 2;

    const float* x   = (const float*)d_in[0];
    const int*   ei  = (const int*)d_in[1];
    const int*   src = ei;
    const int*   dst = ei + E;
    const float* W0  = (const float*)d_in[2];  const float* b0  = (const float*)d_in[3];
    const float* W1  = (const float*)d_in[4];  const float* b1  = (const float*)d_in[5];
    const float* W2  = (const float*)d_in[6];  const float* b2  = (const float*)d_in[7];
    const float* Wfc = (const float*)d_in[8];  const float* bfc = (const float*)d_in[9];

    // ---- workspace carve-up -------------------------------------------------
    char* w = (char*)d_ws;
    size_t off = 0;
    auto carve = [&](size_t bytes) -> void* {
        void* p = w + off;
        off += (bytes + 255) & ~(size_t)255;
        return p;
    };
    float*  hpre = (float*)carve((size_t)N * 128 * sizeof(float));
    float*  agg  = (float*)carve((size_t)N * 128 * sizeof(float));  // reused as logits (ld 48)
    float*  X    = (float*)carve((size_t)N * 128 * sizeof(float));
    float*  dis  = (float*)carve((size_t)N * sizeof(float));
    float*  norm = (float*)carve((size_t)E * sizeof(float));
    __bf16* Wp0  = (__bf16*)carve((size_t)16 * 8 * 512 * sizeof(__bf16)); // K32=16,NT=8
    __bf16* Wp1  = (__bf16*)carve((size_t)4  * 8 * 512 * sizeof(__bf16)); // K32=4, NT=8
    __bf16* Wp2  = (__bf16*)carve((size_t)4  * 8 * 512 * sizeof(__bf16));
    __bf16* Wpfc = (__bf16*)carve((size_t)4  * 3 * 512 * sizeof(__bf16)); // K32=4, NT=3
    (void)ws_size;

    const int B = 256;

    // ---- weights -> bf16 fragment layout -----------------------------------
    repack_w_kernel<<<cdiv(16 * 8 * 512, B), B, 0, stream>>>(W0,  Wp0,  500, 128, 16, 8);
    repack_w_kernel<<<cdiv(4  * 8 * 512, B), B, 0, stream>>>(W1,  Wp1,  128, 128, 4,  8);
    repack_w_kernel<<<cdiv(4  * 8 * 512, B), B, 0, stream>>>(W2,  Wp2,  128, 128, 4,  8);
    repack_w_kernel<<<cdiv(4  * 3 * 512, B), B, 0, stream>>>(Wfc, Wpfc, 128, 40,  4,  3);

    // ---- symmetric norm precompute (shared by all 3 layers) ----------------
    init_deg_kernel<<<cdiv(N, B), B, 0, stream>>>(dis, N);
    deg_accum_kernel<<<cdiv(E, B), B, 0, stream>>>(dst, dis, E);
    rsqrt_kernel<<<cdiv(N, B), B, 0, stream>>>(dis, N);
    edge_norm_kernel<<<cdiv(E, B), B, 0, stream>>>(src, dst, dis, norm, E);

    const int mtiles    = (N + 15) / 16;
    const int gemm_grid = cdiv(mtiles, 8);

    // ---- layer 0: h = x @ W0 ; aggregate ; relu ----------------------------
    gemm_wmma_bf16<8><<<gemm_grid, B, 0, stream>>>(x, Wp0, hpre, N, 500, 500, 16, 128, 128);
    zero4_kernel<<<cdiv(N * 32, B), B, 0, stream>>>((float4*)agg, N * 32);
    scatter_edges_kernel<<<cdiv(E * 32, B), B, 0, stream>>>(hpre, norm, src, dst, agg, E);
    finalize_kernel<<<cdiv(N * 32, B), B, 0, stream>>>((const float4*)agg, (const float4*)hpre,
                                                       dis, b0, (float4*)X, N);

    // ---- layer 1 ------------------------------------------------------------
    gemm_wmma_bf16<8><<<gemm_grid, B, 0, stream>>>(X, Wp1, hpre, N, 128, 128, 4, 128, 128);
    zero4_kernel<<<cdiv(N * 32, B), B, 0, stream>>>((float4*)agg, N * 32);
    scatter_edges_kernel<<<cdiv(E * 32, B), B, 0, stream>>>(hpre, norm, src, dst, agg, E);
    finalize_kernel<<<cdiv(N * 32, B), B, 0, stream>>>((const float4*)agg, (const float4*)hpre,
                                                       dis, b1, (float4*)X, N);

    // ---- layer 2 ------------------------------------------------------------
    gemm_wmma_bf16<8><<<gemm_grid, B, 0, stream>>>(X, Wp2, hpre, N, 128, 128, 4, 128, 128);
    zero4_kernel<<<cdiv(N * 32, B), B, 0, stream>>>((float4*)agg, N * 32);
    scatter_edges_kernel<<<cdiv(E * 32, B), B, 0, stream>>>(hpre, norm, src, dst, agg, E);
    finalize_kernel<<<cdiv(N * 32, B), B, 0, stream>>>((const float4*)agg, (const float4*)hpre,
                                                       dis, b2, (float4*)X, N);

    // ---- FC (N x 128 x 40, ld 48) + log_softmax -----------------------------
    gemm_wmma_bf16<3><<<gemm_grid, B, 0, stream>>>(X, Wpfc, agg, N, 128, 128, 4, 48, 40);
    logsoftmax_kernel<<<cdiv(N, B), B, 0, stream>>>(agg, bfc, (float*)d_out, N);
}
